// LinearAttention_59785944761114
// MI455X (gfx1250) — matrix operand order repaired and checked by reading, not compiled, not run on hardware
//
#include <hip/hip_runtime.h>

// ---------------------------------------------------------------------------
// Types for gfx1250 WMMA (wave32): v_wmma_f32_16x16x32_bf16
// ---------------------------------------------------------------------------
typedef __bf16 bf16_t;
typedef bf16_t v16bf __attribute__((ext_vector_type(16)));
typedef bf16_t v8bf  __attribute__((ext_vector_type(8)));
typedef float  v8f   __attribute__((ext_vector_type(8)));
typedef float  v4f   __attribute__((ext_vector_type(4)));

#define DN 256      // feature dim D
#define NN 4096     // query tokens per batch (== H*W)
#define MN 4096     // kv tokens per batch
#define BN 8        // batch

__device__ __forceinline__ bf16_t f2bf(float x) { return (bf16_t)x; }

__device__ __forceinline__ v8f wmma_bf16(v16bf a, v16bf b, v8f c) {
  return __builtin_amdgcn_wmma_f32_16x16x32_bf16(
      /*neg_a=*/false, a, /*neg_b=*/false, b,
      /*c_mod=*/(short)0, c, /*reuse_a=*/false, /*reuse_b=*/false);
}

// ---------------------------------------------------------------------------
// Operand loaders.
// A operand 16x32 (MxK) per-lane layout: lane (m = lane&15, h = lane>>4) holds
// two contiguous 16B chunks at k = 8h..8h+7 (VGPR0-3) and 16+8h..+7 (VGPR4-7).
// B operand 32x16 (KxN) from transposed storage bT[n][k]: lane (n = lane&15,
// h = lane>>4) holds k = 8h..8h+7 and 16+8h..+7 (same order as A).
// ---------------------------------------------------------------------------
__device__ __forceinline__ v16bf load_A(const bf16_t* __restrict__ base,
                                        int lda, int k0) {
  const int lane = threadIdx.x & 31;
  const int m = lane & 15, h = lane >> 4;
  const bf16_t* p = base + (size_t)m * lda + k0;
  union { v16bf v; v8bf h8[2]; } u;
  u.h8[0] = *(const v8bf*)(p + 8 * h);
  u.h8[1] = *(const v8bf*)(p + 16 + 8 * h);
  return u.v;
}

// A operand built from f32 row-major storage, converted bf16 in-register.
__device__ __forceinline__ v16bf load_A_f32(const float* __restrict__ base,
                                            int lda, int k0) {
  const int lane = threadIdx.x & 31;
  const int m = lane & 15, h = lane >> 4;
  const float* p = base + (size_t)m * lda + k0;
  v4f c0 = *(const v4f*)(p + 8 * h);
  v4f c1 = *(const v4f*)(p + 8 * h + 4);
  v4f c2 = *(const v4f*)(p + 16 + 8 * h);
  v4f c3 = *(const v4f*)(p + 16 + 8 * h + 4);
  union { v16bf v; v8bf h8[2]; } u;
#pragma unroll
  for (int i = 0; i < 4; ++i) {
    u.h8[0][i]     = f2bf(c0[i]);
    u.h8[0][4 + i] = f2bf(c1[i]);
    u.h8[1][i]     = f2bf(c2[i]);
    u.h8[1][4 + i] = f2bf(c3[i]);
  }
  return u.v;
}

__device__ __forceinline__ v16bf load_B(const bf16_t* __restrict__ bT,
                                        int ldb, int k0, int n0) {
  const int lane = threadIdx.x & 31;
  const int n = lane & 15, h = lane >> 4;
  const bf16_t* p = bT + (size_t)(n0 + n) * ldb + k0;
  union { v16bf v; v8bf h8[2]; } u;
  u.h8[0] = *(const v8bf*)(p + 8 * h);
  u.h8[1] = *(const v8bf*)(p + 16 + 8 * h);
  return u.v;
}

// C layout: lane l, VGPR v -> (row m = v + 8*(l>>4), col n = l&15).
__device__ __forceinline__ void store_C_bf16_R(bf16_t* __restrict__ out, int ldo,
                                               int m0, int n0, v8f acc) {
  const int lane = threadIdx.x & 31;
  const int n = lane & 15, h = lane >> 4;
#pragma unroll
  for (int v = 0; v < 8; ++v)
    out[(size_t)(m0 + v + 8 * h) * ldo + n0 + n] = f2bf(acc[v]);
}

// Transposed bf16 store outT[n][m]: 8 rows contiguous in m -> one 16B store.
__device__ __forceinline__ void store_C_bf16_T(bf16_t* __restrict__ outT, int ldo,
                                               int m0, int n0, v8f acc) {
  const int lane = threadIdx.x & 31;
  const int n = lane & 15, h = lane >> 4;
  v8bf pk;
#pragma unroll
  for (int v = 0; v < 8; ++v) pk[v] = f2bf(acc[v]);
  *(v8bf*)(outT + (size_t)(n0 + n) * ldo + m0 + 8 * h) = pk;
}

__device__ __forceinline__ float wave_rowsum_reduce(float s) {
  s += __shfl_xor(s, 1, 32);
  s += __shfl_xor(s, 2, 32);
  s += __shfl_xor(s, 4, 32);
  s += __shfl_xor(s, 8, 32);
  return s;
}

// ---------------------------------------------------------------------------
// Elementwise f32 -> bf16 conversion (weights only)
// ---------------------------------------------------------------------------
__global__ void convert_bf16_kernel(const float* __restrict__ in,
                                    bf16_t* __restrict__ out, int n) {
  int i = blockIdx.x * blockDim.x + threadIdx.x;
  int stride = gridDim.x * blockDim.x;
  for (; i < n; i += stride) out[i] = f2bf(in[i]);
}

// ---------------------------------------------------------------------------
// kv: [B][D][M] f32 (tokens column-major) -> [B][M][D] bf16 (row-major tokens)
// ---------------------------------------------------------------------------
__global__ void transpose_convert_kernel(const float* __restrict__ in,
                                         bf16_t* __restrict__ out) {
  __shared__ float tile[32][33];
  const int b = blockIdx.z;
  const int m0 = blockIdx.x * 32, d0 = blockIdx.y * 32;
  const float* ib = in + ((size_t)b * DN + d0) * MN + m0;
#pragma unroll
  for (int k = 0; k < 4; ++k)
    tile[threadIdx.y + 8 * k][threadIdx.x] = ib[(size_t)(threadIdx.y + 8 * k) * MN + threadIdx.x];
  __syncthreads();
  bf16_t* ob = out + ((size_t)b * MN + m0) * DN + d0;
#pragma unroll
  for (int k = 0; k < 4; ++k)
    ob[(size_t)(threadIdx.y + 8 * k) * DN + threadIdx.x] =
        f2bf(tile[threadIdx.x][threadIdx.y + 8 * k]);
}

// ---------------------------------------------------------------------------
// Q projection: Q = elu(query@Wq^T + bq)+1, qsum rowsum, bf16 row-major out.
// A tile (16x256) hoisted into registers, converted from f32 on the fly.
// grid = (N/16, B), block = 32
// ---------------------------------------------------------------------------
__global__ __launch_bounds__(32)
void qproj_kernel(const float* __restrict__ query, const bf16_t* __restrict__ Wq,
                  const float* __restrict__ bias, bf16_t* __restrict__ Qb,
                  float* __restrict__ qsum) {
  const int lane = threadIdx.x & 31;
  const int n16 = lane & 15, h = lane >> 4;
  const int b = blockIdx.y;
  const int m0 = blockIdx.x * 16;

  const float* Xb = query + ((size_t)b * NN + m0) * DN;
  v16bf a[8];
#pragma unroll
  for (int kc = 0; kc < 8; ++kc) a[kc] = load_A_f32(Xb, DN, kc * 32);

  float rs[8];
#pragma unroll
  for (int v = 0; v < 8; ++v) rs[v] = 0.f;

  for (int dt = 0; dt < DN / 16; ++dt) {
    const int e0 = dt * 16;
    v8f acc;
#pragma unroll
    for (int v = 0; v < 8; ++v) acc[v] = 0.f;
#pragma unroll
    for (int kc = 0; kc < 8; ++kc)
      acc = wmma_bf16(a[kc], load_B(Wq, DN, kc * 32, e0), acc);
    const float bia = bias[e0 + n16];
#pragma unroll
    for (int v = 0; v < 8; ++v) {
      float x = acc[v] + bia;
      x = (x > 0.f) ? (x + 1.f) : __expf(x);  // elu(x)+1
      acc[v] = x;
      rs[v] += x;
    }
    store_C_bf16_R(Qb + (size_t)b * NN * DN, DN, m0, e0, acc);
  }

#pragma unroll
  for (int v = 0; v < 8; ++v) rs[v] = wave_rowsum_reduce(rs[v]);
  if (n16 == 0) {
#pragma unroll
    for (int v = 0; v < 8; ++v) qsum[(size_t)b * NN + m0 + 8 * h + v] = rs[v];
  }
}

// ---------------------------------------------------------------------------
// Fused K+V projection: shares the hoisted kvA A-tile.
//   K = elu(kv@Wk^T+bk)+1 -> Ktb[c][m] (transposed), Z rowsum
//   V = kv@Wv^T+bv        -> Vtb[d][m] (transposed)
// grid = (M/16, B), block = 32
// ---------------------------------------------------------------------------
__global__ __launch_bounds__(32)
void kvproj_kernel(const bf16_t* __restrict__ kvA,
                   const bf16_t* __restrict__ Wk, const float* __restrict__ bk,
                   const bf16_t* __restrict__ Wv, const float* __restrict__ bv,
                   bf16_t* __restrict__ Ktb, bf16_t* __restrict__ Vtb,
                   float* __restrict__ Zs) {
  const int lane = threadIdx.x & 31;
  const int n16 = lane & 15, h = lane >> 4;
  const int b = blockIdx.y;
  const int m0 = blockIdx.x * 16;

  const bf16_t* Xb = kvA + ((size_t)b * MN + m0) * DN;
  v16bf a[8];
#pragma unroll
  for (int kc = 0; kc < 8; ++kc) a[kc] = load_A(Xb, DN, kc * 32);

  float rs[8];
#pragma unroll
  for (int v = 0; v < 8; ++v) rs[v] = 0.f;

  for (int dt = 0; dt < DN / 16; ++dt) {
    const int e0 = dt * 16;
    v8f accK, accV;
#pragma unroll
    for (int v = 0; v < 8; ++v) { accK[v] = 0.f; accV[v] = 0.f; }
#pragma unroll
    for (int kc = 0; kc < 8; ++kc) {
      accK = wmma_bf16(a[kc], load_B(Wk, DN, kc * 32, e0), accK);
      accV = wmma_bf16(a[kc], load_B(Wv, DN, kc * 32, e0), accV);
    }
    const float biaK = bk[e0 + n16];
    const float biaV = bv[e0 + n16];
#pragma unroll
    for (int v = 0; v < 8; ++v) {
      float x = accK[v] + biaK;
      x = (x > 0.f) ? (x + 1.f) : __expf(x);  // elu(x)+1
      accK[v] = x;
      rs[v] += x;
      accV[v] += biaV;
    }
    store_C_bf16_T(Ktb + (size_t)b * DN * MN, MN, m0, e0, accK);
    store_C_bf16_T(Vtb + (size_t)b * DN * MN, MN, m0, e0, accV);
  }

#pragma unroll
  for (int v = 0; v < 8; ++v) rs[v] = wave_rowsum_reduce(rs[v]);
  if (n16 == 0) {
#pragma unroll
    for (int v = 0; v < 8; ++v) Zs[(size_t)b * MN + m0 + 8 * h + v] = rs[v];
  }
}

// ---------------------------------------------------------------------------
// KV[c][d] = sum_m K[m][c]*V[m][d] via Ktb[c][m], Vtb[d][m]; out KVt[d][c].
// grid = (16, 16, B), block = 32
// ---------------------------------------------------------------------------
__global__ __launch_bounds__(32)
void kv_accum_kernel(const bf16_t* __restrict__ Ktb, const bf16_t* __restrict__ Vtb,
                     bf16_t* __restrict__ KVt) {
  const int b = blockIdx.z;
  const int c0 = blockIdx.x * 16, d0 = blockIdx.y * 16;
  const bf16_t* Kb = Ktb + (size_t)b * DN * MN + (size_t)c0 * MN;
  const bf16_t* Vb = Vtb + (size_t)b * DN * MN;
  v8f acc;
#pragma unroll
  for (int v = 0; v < 8; ++v) acc[v] = 0.f;
  for (int mc = 0; mc < MN / 32; ++mc) {
    v16bf a = load_A(Kb, MN, mc * 32);       // A[c][m]
    v16bf bm = load_B(Vb, MN, mc * 32, d0);  // B[m][d] from Vtb[d][m]
    acc = wmma_bf16(a, bm, acc);
  }
  store_C_bf16_T(KVt + (size_t)b * DN * DN, DN, c0, d0, acc);
}

// ---------------------------------------------------------------------------
// Fused output stage (per 16-token strip, one wave):
//   num = Q @ KV ; out = num / (qsum*Z + 1e-6) -> LDS (bf16 strip, 8 KB)
//   final = out @ Wo^T + bo  -> f32 d_out
// LDS store->load within a single wave is ordered by DScnt (no barrier).
// grid = (N/16, B), block = 32
// ---------------------------------------------------------------------------
__global__ __launch_bounds__(32)
void out_fused_kernel(const bf16_t* __restrict__ Qb, const bf16_t* __restrict__ KVt,
                      const float* __restrict__ qsum, const float* __restrict__ Zs,
                      const bf16_t* __restrict__ Wo, const float* __restrict__ bo,
                      float* __restrict__ out) {
  __shared__ __align__(16) bf16_t strip[16 * DN];  // [16 tokens][256] bf16
  const int lane = threadIdx.x & 31;
  const int n16 = lane & 15, h = lane >> 4;
  const int b = blockIdx.y;
  const int m0 = blockIdx.x * 16;

  // ---- phase 1: num = Q @ KV, normalized, into LDS ----
  const bf16_t* Qbase = Qb + ((size_t)b * NN + m0) * DN;
  v16bf a[8];
#pragma unroll
  for (int kc = 0; kc < 8; ++kc) a[kc] = load_A(Qbase, DN, kc * 32);

  const float* qs = qsum + (size_t)b * NN + m0 + 8 * h;
  const float* zz = Zs + (size_t)b * NN + m0 + 8 * h;
  float rden[8];
#pragma unroll
  for (int v = 0; v < 8; ++v) rden[v] = 1.f / (qs[v] * zz[v] + 1e-6f);

  const bf16_t* KVb = KVt + (size_t)b * DN * DN;
  for (int dt = 0; dt < DN / 16; ++dt) {
    const int e0 = dt * 16;
    v8f acc;
#pragma unroll
    for (int v = 0; v < 8; ++v) acc[v] = 0.f;
#pragma unroll
    for (int kc = 0; kc < 8; ++kc)
      acc = wmma_bf16(a[kc], load_B(KVb, DN, kc * 32, e0), acc);
#pragma unroll
    for (int v = 0; v < 8; ++v)
      strip[(size_t)(v + 8 * h) * DN + e0 + n16] = f2bf(acc[v] * rden[v]);
  }

  // ---- phase 2: final = strip @ Wo^T + bo (A re-hoisted from LDS) ----
  v16bf a2[8];
#pragma unroll
  for (int kc = 0; kc < 8; ++kc) {
    const bf16_t* p = strip + (size_t)n16 * DN + kc * 32;
    union { v16bf v; v8bf h8[2]; } u;
    u.h8[0] = *(const v8bf*)(p + 8 * h);
    u.h8[1] = *(const v8bf*)(p + 16 + 8 * h);
    a2[kc] = u.v;
  }

  float* ob = out + ((size_t)b * NN + m0) * DN;
  for (int dt = 0; dt < DN / 16; ++dt) {
    const int e0 = dt * 16;
    v8f acc;
#pragma unroll
    for (int v = 0; v < 8; ++v) acc[v] = 0.f;
#pragma unroll
    for (int kc = 0; kc < 8; ++kc)
      acc = wmma_bf16(a2[kc], load_B(Wo, DN, kc * 32, e0), acc);
    const float bia = bo[e0 + n16];
#pragma unroll
    for (int v = 0; v < 8; ++v)
      ob[(size_t)(v + 8 * h) * DN + e0 + n16] = acc[v] + bia;
  }
}

// ---------------------------------------------------------------------------
// Host-side orchestration
// ---------------------------------------------------------------------------
extern "C" void kernel_launch(void* const* d_in, const int* in_sizes, int n_in,
                              void* d_out, int out_size, void* d_ws, size_t ws_size,
                              hipStream_t stream) {
  const float* query = (const float*)d_in[0];  // [B,N,D]
  const float* keyv  = (const float*)d_in[1];  // [B,D,H,W]
  const float* Wq = (const float*)d_in[2];
  const float* bq = (const float*)d_in[3];
  const float* Wk = (const float*)d_in[4];
  const float* bk = (const float*)d_in[5];
  const float* Wv = (const float*)d_in[6];
  const float* bv = (const float*)d_in[7];
  const float* Wo = (const float*)d_in[8];
  const float* bo = (const float*)d_in[9];
  float* out = (float*)d_out;

  const size_t szW   = (size_t)DN * DN * sizeof(bf16_t);       // 128 KB
  const size_t szTok = (size_t)BN * NN * DN * sizeof(bf16_t);  // 16 MB
  const size_t szSum = (size_t)BN * NN * sizeof(float);        // 128 KB
  const size_t szKV  = (size_t)BN * DN * DN * sizeof(bf16_t);  // 1 MB

  char* w = (char*)d_ws;
  bf16_t* Wq_b = (bf16_t*)w; w += szW;
  bf16_t* Wk_b = (bf16_t*)w; w += szW;
  bf16_t* Wv_b = (bf16_t*)w; w += szW;
  bf16_t* Wo_b = (bf16_t*)w; w += szW;
  bf16_t* kvA  = (bf16_t*)w; w += szTok;  // kv tokens row-major bf16
  bf16_t* Qb   = (bf16_t*)w; w += szTok;  // elu(Q)+1
  bf16_t* Ktb  = (bf16_t*)w; w += szTok;  // K transposed [B][D][M]
  bf16_t* Vtb  = (bf16_t*)w; w += szTok;  // V transposed [B][D][M]
  bf16_t* KVt  = (bf16_t*)w; w += szKV;   // KV transposed [B][D][D]
  float*  qsum = (float*)w;  w += szSum;
  float*  Zs   = (float*)w;  w += szSum;

  // 1) convert weights to bf16
  convert_bf16_kernel<<<64, 256, 0, stream>>>(Wq, Wq_b, DN * DN);
  convert_bf16_kernel<<<64, 256, 0, stream>>>(Wk, Wk_b, DN * DN);
  convert_bf16_kernel<<<64, 256, 0, stream>>>(Wv, Wv_b, DN * DN);
  convert_bf16_kernel<<<64, 256, 0, stream>>>(Wo, Wo_b, DN * DN);

  // 2) kv [B,D,H,W] -> row-major bf16 tokens [B,M,D]
  transpose_convert_kernel<<<dim3(MN / 32, DN / 32, BN), dim3(32, 8), 0, stream>>>(keyv, kvA);

  // 3) Q projection (f32 input converted in-register) + qsum
  qproj_kernel<<<dim3(NN / 16, BN), 32, 0, stream>>>(query, Wq_b, bq, Qb, qsum);

  // 4) fused K+V projections (+Z), transposed outputs
  kvproj_kernel<<<dim3(MN / 16, BN), 32, 0, stream>>>(kvA, Wk_b, bk, Wv_b, bv,
                                                      Ktb, Vtb, Zs);

  // 5) KV = K^T V (stored transposed)
  kv_accum_kernel<<<dim3(DN / 16, DN / 16, BN), 32, 0, stream>>>(Ktb, Vtb, KVt);

  // 6) fused: num -> normalize -> output projection (f32 out)
  out_fused_kernel<<<dim3(NN / 16, BN), 32, 0, stream>>>(Qb, KVt, qsum, Zs,
                                                         Wo_b, bo, out);

  (void)in_sizes; (void)n_in; (void)out_size; (void)ws_size;
}
